// CascadedPerceiverIO_3375844294754
// MI455X (gfx1250) — compile-verified, matrix operand order checked by reading, not run on hardware
//
#include <hip/hip_runtime.h>
#include <hip/hip_bf16.h>
#include <math.h>

// ---------------------------------------------------------------------------
// MI455X (gfx1250) implementation of CascadedPerceiverIO forward.
// - GEMMs on v_wmma_f32_16x16x32_bf16 (bf16 operands, fp32 accumulate).
// - GEMM tiles staged LDS-side by the Tensor Data Mover
//   (tensor_load_to_lds + s_wait_tensorcnt), with TDM OOB zero-fill
//   replacing all edge guards.
// - Weight transpose/convert kernel stages via global_load_async_to_lds_b128
//   (ASYNCcnt path).
// Workload is HBM-bandwidth bound (~1 GB traffic, ~0.3 TFLOP) -> bf16 halves
// GEMM operand bytes; fp32 accum keeps accuracy near the fp32 reference.
// ---------------------------------------------------------------------------

typedef __attribute__((ext_vector_type(16))) __bf16 v16bf;
typedef __attribute__((ext_vector_type(8)))  __bf16 v8bf;
typedef __attribute__((ext_vector_type(4)))  __bf16 v4bf;
typedef __attribute__((ext_vector_type(8)))  float  v8f;
typedef __attribute__((ext_vector_type(4)))  unsigned int u32x4;
typedef __attribute__((ext_vector_type(8)))  int i32x8;
typedef __attribute__((ext_vector_type(4)))  int i32x4;

#define BATCH  4
#define NTOK   8192
#define NQTOK  8192
#define DLAT   128
#define NLAT   128
#define INDIM  128
#define QDIM   96
#define CH_    4
#define CDH_   128
#define SH_    8
#define SDH_   128

// ---------------------------------------------------------------------------
// f32 -> bf16 conversion (plain)
// ---------------------------------------------------------------------------
__global__ void f2bf_kernel(const float* __restrict__ in, __bf16* __restrict__ out, long n) {
    long i = ((long)blockIdx.x * blockDim.x + threadIdx.x) * 4;
    if (i + 4 <= n) {
        float4 v = *(const float4*)(in + i);
        v4bf o;
        o[0] = (__bf16)v.x; o[1] = (__bf16)v.y; o[2] = (__bf16)v.z; o[3] = (__bf16)v.w;
        *(v4bf*)(out + i) = o;
    } else {
        for (long j = i; j < n; ++j) out[j] = (__bf16)in[j];
    }
}

// ---------------------------------------------------------------------------
// f32 [K][N] -> bf16 [N][K] transposing conversion. Interior 32x32 tiles are
// staged into LDS with async global->LDS copies (ASYNCcnt), edges fall back
// to guarded VGPR loads. blockIdx.z = batch (packed K*N per batch).
// ---------------------------------------------------------------------------
__global__ __launch_bounds__(256)
void f2bfT_kernel(const float* __restrict__ in, __bf16* __restrict__ out, int K, int N) {
    __shared__ float tile[32][36];   // 36 floats = 144B row stride (16B aligned)
    const long zo = (long)blockIdx.z * K * N;
    in += zo; out += zo;
    const int k0 = blockIdx.y * 32, n0 = blockIdx.x * 32;
    const int tx = threadIdx.x & 31, ty = threadIdx.x >> 5;

    const bool interior = (k0 + 32 <= K) && (n0 + 32 <= N);
    if (interior) {
        // 256 threads x 16B: thread t copies in[k0 + (t>>3)][n0 + (t&7)*4 ..+3]
        const int r = threadIdx.x >> 3, c = threadIdx.x & 7;
        const float* src = in + (long)(k0 + r) * N + n0 + c * 4;
        unsigned ldsOff = (unsigned)(unsigned long long)(size_t)&tile[r][c * 4];
        asm volatile("global_load_async_to_lds_b128 %0, %1, off"
                     :: "v"(ldsOff), "v"(src) : "memory");
        asm volatile("s_wait_asynccnt 0" ::: "memory");
    } else {
        for (int r = ty; r < 32; r += 8) {
            const int k = k0 + r, n = n0 + tx;
            tile[r][tx] = (k < K && n < N) ? in[(long)k * N + n] : 0.f;
        }
    }
    __syncthreads();
    for (int r = ty; r < 32; r += 8) {
        const int n = n0 + r, k = k0 + tx;
        if (n < N && k < K) out[(long)n * K + k] = (__bf16)tile[tx][r];
    }
}

// ---------------------------------------------------------------------------
// Tensor Data Mover: DMA a 128x32 bf16 2D tile (row stride ldk elements) from
// global into LDS. tensor_dim = remaining extent -> TDM zero-fills OOB.
// Descriptor packing per CDNA5 ISA 8.3/8.4 (D# groups 0/1; groups 2/3 zero).
// ---------------------------------------------------------------------------
__device__ __forceinline__ void dset64(unsigned long long* q, int pos, unsigned long long val) {
    const int w = pos >> 6, b = pos & 63;
    q[w] |= val << b;
    if (b) q[w + 1] |= val >> (64 - b);
}

__device__ __forceinline__ void tdm_load_tile_2d(unsigned ldsOff, const __bf16* gptr,
                                                 int ldk, int remRows, int remK) {
    unsigned long long g0q[2] = {0ull, 0ull};
    unsigned long long g1q[4] = {0ull, 0ull, 0ull, 0ull};
    const unsigned long long ga = (unsigned long long)(size_t)gptr;
    g0q[0] = 1ull                                   // count = 1 valid descriptor
           | ((unsigned long long)ldsOff << 32);    // lds_addr
    g0q[1] = (ga & 0x01FFFFFFFFFFFFFFull)           // global_addr[56:0]
           | (2ull << 62);                          // type = image
    dset64(g1q, 16, 1ull);                          // data_size = 2 bytes
    dset64(g1q, 48, (unsigned)remK);                // tensor_dim0 (K extent)
    dset64(g1q, 80, (unsigned)remRows);             // tensor_dim1 (row extent)
    dset64(g1q, 112, 32ull);                        // tile_dim0 = 32 elements
    dset64(g1q, 128, 128ull);                       // tile_dim1 = 128 rows
    dset64(g1q, 160, (unsigned long long)(unsigned)ldk);  // tensor_dim0_stride
    u32x4 g0 = { (unsigned)g0q[0], (unsigned)(g0q[0] >> 32),
                 (unsigned)g0q[1], (unsigned)(g0q[1] >> 32) };
    i32x8 g1 = { (int)(unsigned)g1q[0], (int)(g1q[0] >> 32),
                 (int)(unsigned)g1q[1], (int)(g1q[1] >> 32),
                 (int)(unsigned)g1q[2], (int)(g1q[2] >> 32),
                 (int)(unsigned)g1q[3], (int)(g1q[3] >> 32) };
    i32x4 z4 = { 0, 0, 0, 0 };
    i32x8 z8 = { 0, 0, 0, 0, 0, 0, 0, 0 };
    __builtin_amdgcn_tensor_load_to_lds(g0, g1, z4, z4, z8, 0);
}

// ---------------------------------------------------------------------------
// WMMA GEMM: C[M,N] = A[M,K] x B[N,K]^T (+bias). A row-major [M][K] bf16,
// B row-major [N][K] bf16 (pre-transposed by f2bfT for weights). Block tile
// 128x128, K-step 32, 8 waves (2x4), wave tile 64x32 (4x2 wmma). Both LDS
// tiles staged by the TDM (wave 0 issues, TENSORcnt-waited), double-buffered.
// ---------------------------------------------------------------------------
__device__ __forceinline__ v16bf ld_frag(const __bf16* tile, int lane) {
    // row-major [16][32] tile; ISA 16-bit A/B-frag layout: lane<16 holds
    // K{0..7,16..23} of row=lane&15, lane>=16 holds K{8..15,24..31}.
    const int r = lane & 15, hf = lane >> 4;
    const __bf16* p = tile + r * 32;
    v8bf lo = *(const v8bf*)(p + hf * 8);
    v8bf hi = *(const v8bf*)(p + 16 + hf * 8);
    v16bf o;
#pragma unroll
    for (int i = 0; i < 8; ++i) { o[i] = lo[i]; o[i + 8] = hi[i]; }
    return o;
}

__global__ __launch_bounds__(256)
void wmma_gemm_bf16_kernel(const __bf16* __restrict__ A, const __bf16* __restrict__ Bm,
                           const float* __restrict__ bias, float* __restrict__ C,
                           int M, int N, int K, long sA, long sB, long sC) {
    __shared__ __bf16 lsA[2][128 * 32];
    __shared__ __bf16 lsB[2][128 * 32];
    const int tid = threadIdx.x, lane = tid & 31, wave = tid >> 5;
    const int wm = wave >> 2;   // 0..1 (64 rows each)
    const int wn = wave & 3;    // 0..3 (32 cols each)
    const int m0 = blockIdx.y * 128, n0 = blockIdx.x * 128;
    A  += (long)blockIdx.z * sA;
    Bm += (long)blockIdx.z * sB;
    C  += (long)blockIdx.z * sC;

    v8f acc[4][2] = {};

    auto issueTile = [&](int buf, int kt) {
        if (wave == 0) {   // one wave posts both DMA descriptors
            tdm_load_tile_2d((unsigned)(unsigned long long)(size_t)&lsA[buf][0],
                             A + (long)m0 * K + kt, K, M - m0, K - kt);
            tdm_load_tile_2d((unsigned)(unsigned long long)(size_t)&lsB[buf][0],
                             Bm + (long)n0 * K + kt, K, N - n0, K - kt);
        }
    };

    auto compute = [&](int buf) {
        v16bf aF[4], bF[2];
#pragma unroll
        for (int tm = 0; tm < 4; ++tm)
            aF[tm] = ld_frag(&lsA[buf][(wm * 64 + tm * 16) * 32], lane);
#pragma unroll
        for (int tn = 0; tn < 2; ++tn)
            bF[tn] = ld_frag(&lsB[buf][(wn * 32 + tn * 16) * 32], lane);
#pragma unroll
        for (int tm = 0; tm < 4; ++tm)
#pragma unroll
            for (int tn = 0; tn < 2; ++tn)
                acc[tm][tn] = __builtin_amdgcn_wmma_f32_16x16x32_bf16(
                    false, aF[tm], false, bF[tn], (short)0, acc[tm][tn], false, false);
    };

    const int nk = (K + 31) / 32;
    issueTile(0, 0);                                    // TENSORcnt = 2
    for (int t = 0; t < nk; ++t) {
        const int cur = t & 1;
        const bool more = (t + 1 < nk);
        if (more) issueTile(cur ^ 1, (t + 1) * 32);     // TENSORcnt = 4
        if (wave == 0) {
            if (more) __builtin_amdgcn_s_wait_tensorcnt((short)2);  // cur tile done
            else      __builtin_amdgcn_s_wait_tensorcnt((short)0);
        }
        __syncthreads();        // release all waves to consume cur
        compute(cur);
        __syncthreads();        // everyone done with cur before it is re-DMAed
    }

    // Epilogue: C/D layout -> lane<16: M=r, lane>=16: M=8+r; N=lane&15.
    const int hf = lane >> 4, lx = lane & 15;
#pragma unroll
    for (int tm = 0; tm < 4; ++tm) {
#pragma unroll
        for (int tn = 0; tn < 2; ++tn) {
            const int ng = n0 + wn * 32 + tn * 16 + lx;
            const float bv = (bias && ng < N) ? bias[ng] : 0.f;
#pragma unroll
            for (int r = 0; r < 8; ++r) {
                const int mg = m0 + wm * 64 + tm * 16 + hf * 8 + r;
                if (mg < M && ng < N)
                    C[(long)mg * N + ng] = acc[tm][tn][r] + bv;
            }
        }
    }
}

// ---------------------------------------------------------------------------
// Fused elementwise / reduction kernels (wave32-aware)
// ---------------------------------------------------------------------------
__global__ void ln_kernel(const float* __restrict__ x, const float* __restrict__ gw,
                          const float* __restrict__ bw, float* __restrict__ y,
                          int R, int D) {
    const int row = blockIdx.x * 8 + (threadIdx.x >> 5);
    const int lane = threadIdx.x & 31;
    if (row >= R) return;
    const float* xr = x + (long)row * D;
    float s = 0.f, s2 = 0.f;
    for (int i = lane; i < D; i += 32) { float v = xr[i]; s += v; s2 += v * v; }
#pragma unroll
    for (int o = 16; o > 0; o >>= 1) { s += __shfl_xor(s, o, 32); s2 += __shfl_xor(s2, o, 32); }
    const float mean = s / D;
    const float inv = rsqrtf(fmaxf(s2 / D - mean * mean, 0.f) + 1e-5f);
    float* yr = y + (long)row * D;
    for (int i = lane; i < D; i += 32) yr[i] = (xr[i] - mean) * inv * gw[i] + bw[i];
}

__global__ void softmax_kernel(float* __restrict__ x, int R, int Ncol, float scale) {
    const int row = blockIdx.x * 8 + (threadIdx.x >> 5);
    const int lane = threadIdx.x & 31;
    if (row >= R) return;
    float* xr = x + (long)row * Ncol;
    float mx = -3.4e38f;
    for (int i = lane; i < Ncol; i += 32) mx = fmaxf(mx, xr[i] * scale);
#pragma unroll
    for (int o = 16; o > 0; o >>= 1) mx = fmaxf(mx, __shfl_xor(mx, o, 32));
    float s = 0.f;
    for (int i = lane; i < Ncol; i += 32) { float e = expf(xr[i] * scale - mx); xr[i] = e; s += e; }
#pragma unroll
    for (int o = 16; o > 0; o >>= 1) s += __shfl_xor(s, o, 32);
    const float inv = 1.f / s;
    for (int i = lane; i < Ncol; i += 32) xr[i] *= inv;
}

__device__ __forceinline__ float gelu_exact(float v) {
    return 0.5f * v * (1.f + erff(v * 0.70710678118654752f));
}

__global__ void gelu_kernel(float* __restrict__ y, const float* __restrict__ x, long n) {
    long i = (long)blockIdx.x * blockDim.x + threadIdx.x;
    if (i < n) y[i] = gelu_exact(x[i]);
}

__global__ void geglu_kernel(const float* __restrict__ hid, float* __restrict__ out,
                             long R, int H) {   // hid: [R, 2H] -> out: [R, H]
    long i = (long)blockIdx.x * blockDim.x + threadIdx.x;
    if (i >= R * (long)H) return;
    const long r = i / H; const int c = (int)(i % H);
    const float a = hid[r * 2 * H + c];
    const float g = hid[r * 2 * H + H + c];
    out[i] = a * gelu_exact(g);
}

__global__ void add_kernel(float* __restrict__ y, const float* __restrict__ a,
                           const float* __restrict__ b, long n) {
    long i = (long)blockIdx.x * blockDim.x + threadIdx.x;
    if (i < n) y[i] = a[i] + b[i];
}

// in: [B, Nn, (ldin cols)] taking h*dh slice at col `off` -> out: [B*h, Nn, dh]
__global__ void split_heads_kernel(const float* __restrict__ in, float* __restrict__ out,
                                   int Nn, int h, int dh, int ldin, int off) {
    long i = (long)blockIdx.x * blockDim.x + threadIdx.x;
    const long total = (long)BATCH * h * Nn * dh;
    if (i >= total) return;
    const int d = (int)(i % dh); long t = i / dh;
    const int n = (int)(t % Nn); t /= Nn;
    const int hh = (int)(t % h); const int b = (int)(t / h);
    out[i] = in[((long)(b * Nn + n)) * ldin + off + hh * dh + d];
}

// in: [B*h, Nn, dh] -> out: [B, Nn, h*dh]
__global__ void merge_heads_kernel(const float* __restrict__ in, float* __restrict__ out,
                                   int Nn, int h, int dh) {
    long i = (long)blockIdx.x * blockDim.x + threadIdx.x;
    const long total = (long)BATCH * h * Nn * dh;
    if (i >= total) return;
    const int d = (int)(i % dh); long t = i / dh;
    const int hh = (int)(t % h); t /= h;
    const int n = (int)(t % Nn); const int b = (int)(t / Nn);
    out[i] = in[((long)(b * h + hh) * Nn + n) * dh + d];
}

__global__ void mean_rows_kernel(const float* __restrict__ g, float* __restrict__ out,
                                 int T, int D) {   // g: [B, T, D] -> out: [B, D]
    const int i = blockIdx.x * blockDim.x + threadIdx.x;
    if (i >= BATCH * D) return;
    const int b = i / D, d = i % D;
    float s = 0.f;
    for (int t = 0; t < T; ++t) s += g[((long)(b * T + t)) * D + d];
    out[i] = s / T;
}

__global__ void bcast_lat_kernel(const float* __restrict__ latents, float* __restrict__ lat) {
    long i = (long)blockIdx.x * blockDim.x + threadIdx.x;
    if (i < (long)BATCH * NLAT * DLAT) lat[i] = latents[i % (NLAT * DLAT)];
}

__global__ void concat_lat_kernel(const float* __restrict__ lat, float* __restrict__ fused, int m) {
    long i = (long)blockIdx.x * blockDim.x + threadIdx.x;
    if (i >= (long)BATCH * NLAT * DLAT) return;
    const int d = (int)(i % DLAT); long t = i / DLAT;
    const int row = (int)(t % NLAT); const int b = (int)(t / NLAT);
    fused[((long)b * (3 * NLAT) + m * NLAT + row) * DLAT + d] = lat[i];
}

__global__ void head_kernel(const float* __restrict__ x, const float* __restrict__ w,
                            const float* __restrict__ b, float* __restrict__ y,
                            int M, int K) {
    const int row = blockIdx.x * 8 + (threadIdx.x >> 5);
    const int lane = threadIdx.x & 31;
    if (row >= M) return;
    const float* xr = x + (long)row * K;
    float s = 0.f;
    for (int i = lane; i < K; i += 32) s += xr[i] * w[i];
#pragma unroll
    for (int o = 16; o > 0; o >>= 1) s += __shfl_xor(s, o, 32);
    if (lane == 0) y[row] = s + b[0];
}

// ---------------------------------------------------------------------------
// Host orchestration
// ---------------------------------------------------------------------------
struct Bump {
    char* base; size_t off, cap, floor_;
    void* take(size_t bytes) {
        size_t a = (off + 255) & ~(size_t)255;
        if (a + bytes > cap) a = floor_;          // wrap (stream-serialized, stays in-bounds)
        off = a + bytes;
        return base + a;
    }
};

struct LNp { const float *g, *b; };
struct ATp { const float *wq, *wkv, *wo, *bo; };
struct FFp { const float *w1, *b1, *w2, *b2; };

static inline unsigned gblk(long n, int per) { return (unsigned)((n + per - 1) / per); }

static void conv_bf16(hipStream_t st, const float* in, __bf16* out, long n) {
    f2bf_kernel<<<gblk(n, 1024), 256, 0, st>>>(in, out, n);
}

// C[M,N] = A[M,K] * B + bias.
// bIsNK: B source already stored [N][K] (row-major) -> plain convert.
// else : B source stored [K][N] -> transposing convert to [N][K].
static void gemm(hipStream_t st, Bump bp, const float* A, const float* Bw,
                 const float* bias, float* C, long M, int N, int K,
                 bool bIsNK, int batch) {
    const long eA = (long)batch * M * K;
    const long eB = (long)batch * N * K;
    __bf16* Ab = (__bf16*)bp.take((size_t)eA * 2);
    __bf16* Bb = (__bf16*)bp.take((size_t)eB * 2);
    conv_bf16(st, A, Ab, eA);
    if (bIsNK) {
        conv_bf16(st, Bw, Bb, eB);
    } else {
        dim3 tg((N + 31) / 32, (K + 31) / 32, batch);
        f2bfT_kernel<<<tg, 256, 0, st>>>(Bw, Bb, K, N);
    }
    dim3 grid((N + 127) / 128, (unsigned)((M + 127) / 128), batch);
    wmma_gemm_bf16_kernel<<<grid, 256, 0, st>>>(Ab, Bb, bias, C,
        (int)M, N, K, M * K, (long)N * K, M * N);
}

static void ln(hipStream_t st, const float* x, LNp p, float* y, long R, int D) {
    ln_kernel<<<gblk(R, 8), 256, 0, st>>>(x, p.g, p.b, y, (int)R, D);
}

static void addv(hipStream_t st, float* y, const float* a, const float* b, long n) {
    add_kernel<<<gblk(n, 256), 256, 0, st>>>(y, a, b, n);
}

// Multi-head attention, per reference. xq:[B*Nq,dq] (pre-LN), ctx:[B*Nc,dc].
static void run_attn(hipStream_t st, Bump bp, const float* xq, long Nq, int dq,
                     const float* ctx, long Nc, int dc, int h, int dh,
                     const ATp& p, float* out) {
    const int inner = h * dh;
    const long Mq = (long)BATCH * Nq, Mc = (long)BATCH * Nc;
    float* q = (float*)bp.take((size_t)Mq * inner * 4);
    gemm(st, bp, xq, p.wq, nullptr, q, Mq, inner, dq, false, 1);
    float* kv = (float*)bp.take((size_t)Mc * inner * 2 * 4);
    gemm(st, bp, ctx, p.wkv, nullptr, kv, Mc, 2 * inner, dc, false, 1);

    const long nh = Mq * inner, nhc = Mc * inner;
    float* qh = (float*)bp.take((size_t)nh * 4);
    float* kh = (float*)bp.take((size_t)nhc * 4);
    float* vh = (float*)bp.take((size_t)nhc * 4);
    split_heads_kernel<<<gblk(nh, 256), 256, 0, st>>>(q, qh, (int)Nq, h, dh, inner, 0);
    split_heads_kernel<<<gblk(nhc, 256), 256, 0, st>>>(kv, kh, (int)Nc, h, dh, 2 * inner, 0);
    split_heads_kernel<<<gblk(nhc, 256), 256, 0, st>>>(kv, vh, (int)Nc, h, dh, 2 * inner, inner);

    float* sim = (float*)bp.take((size_t)BATCH * h * Nq * Nc * 4);
    // q.k^T: B operand is kh stored [Nc][dh] == [N][K] already.
    gemm(st, bp, qh, kh, nullptr, sim, Nq, (int)Nc, dh, true, BATCH * h);
    const long rows = (long)BATCH * h * Nq;
    softmax_kernel<<<gblk(rows, 8), 256, 0, st>>>(sim, (int)rows, (int)Nc,
                                                  1.f / sqrtf((float)dh));
    float* oh = (float*)bp.take((size_t)nh * 4);
    // a.v: B operand vh stored [Nc][dh] == [K][N] -> transpose-convert.
    gemm(st, bp, sim, vh, nullptr, oh, Nq, dh, (int)Nc, false, BATCH * h);
    float* om = (float*)bp.take((size_t)nh * 4);
    merge_heads_kernel<<<gblk(nh, 256), 256, 0, st>>>(oh, om, (int)Nq, h, dh);
    gemm(st, bp, om, p.wo, p.bo, out, Mq, dq, inner, false, 1);
}

// GEGLU feed-forward: out = (a * gelu(g)) @ w2 + b2
static void run_ff(hipStream_t st, Bump bp, const float* xln, const FFp& p,
                   long R, int d, float* out) {
    float* hid = (float*)bp.take((size_t)R * d * 8 * 4);
    gemm(st, bp, xln, p.w1, p.b1, hid, R, d * 8, d, false, 1);
    float* gg = (float*)bp.take((size_t)R * d * 4 * 4);
    geglu_kernel<<<gblk(R * (long)d * 4, 256), 256, 0, st>>>(hid, gg, R, d * 4);
    gemm(st, bp, gg, p.w2, p.b2, out, R, d, d * 4, false, 1);
}

struct Blk { const float* latents; LNp ca_ln, ca_lnc; ATp ca; LNp sa_ln; ATp sa; LNp ff_ln; FFp ff; };
struct G2L { const float *w, *b; };
struct Trk { LNp sa_ln; ATp sa; LNp ff_ln; FFp ff; };
struct Dec { LNp ca_ln, ca_lnc; ATp ca; LNp ff_ln; FFp ff; const float *hw, *hb; };

extern "C" void kernel_launch(void* const* d_in, const int* in_sizes, int n_in,
                              void* d_out, int out_size, void* d_ws, size_t ws_size,
                              hipStream_t stream) {
    (void)in_sizes; (void)n_in; (void)out_size;
    const float* xT = (const float*)d_in[0];
    const float* xQ = (const float*)d_in[1];
    const float* xV = (const float*)d_in[2];
    const float* queries = (const float*)d_in[3];

    // Params flattened in setup_inputs() insertion order after the 4 arrays.
    int pc = 4;
    auto P = [&]() { return (const float*)d_in[pc++]; };

    struct Inp { const float *w1, *b1, *w2, *b2; } inp[3];
    for (int m = 0; m < 3; ++m) inp[m] = { P(), P(), P(), P() };

    Blk enc[3][3];
    for (int m = 0; m < 3; ++m)
        for (int s = 0; s < 3; ++s) {
            Blk& e = enc[m][s];
            e.latents = P();
            e.ca_ln  = { P(), P() };
            e.ca_lnc = { P(), P() };
            e.ca     = { P(), P(), P(), P() };
            e.sa_ln  = { P(), P() };
            e.sa     = { P(), P(), P(), P() };
            e.ff_ln  = { P(), P() };
            e.ff     = { P(), P(), P(), P() };
        }
    G2L g2l[3][3];
    for (int m = 0; m < 3; ++m)
        for (int s = 0; s < 3; ++s) g2l[m][s] = { P(), P() };
    Trk trunk[3];
    for (int t = 0; t < 3; ++t) {
        trunk[t].sa_ln = { P(), P() };
        trunk[t].sa    = { P(), P(), P(), P() };
        trunk[t].ff_ln = { P(), P() };
        trunk[t].ff    = { P(), P(), P(), P() };
    }
    Dec dec[3];
    for (int m = 0; m < 3; ++m) {
        dec[m].ca_ln  = { P(), P() };
        dec[m].ca_lnc = { P(), P() };
        dec[m].ca     = { P(), P(), P(), P() };
        dec[m].ff_ln  = { P(), P() };
        dec[m].ff     = { P(), P(), P(), P() };
        dec[m].hw = P(); dec[m].hb = P();
    }

    Bump bp{ (char*)d_ws, 0, ws_size, 0 };
    const long MT  = (long)BATCH * NTOK;          // 32768 token rows
    const long MQ  = (long)BATCH * NQTOK;         // 32768 query rows
    const long MLT = (long)BATCH * NLAT;          // 512 latent rows
    const long TFU = 3 * NLAT;                    // 384 fused tokens

    // Persistent buffers
    float* xs[3];
    for (int m = 0; m < 3; ++m) xs[m] = (float*)bp.take((size_t)MT * INDIM * 4);
    float* gbuf[2];
    gbuf[0] = (float*)bp.take((size_t)BATCH * TFU * DLAT * 4);
    gbuf[1] = (float*)bp.take((size_t)BATCH * TFU * DLAT * 4);
    float* lat = (float*)bp.take((size_t)MLT * DLAT * 4);
    bp.floor_ = bp.off;   // transient region starts here

    // ---- input embeddings: gelu(x @ w1 + b1) @ w2 + b2 ----
    const float* xin[3] = { xT, xQ, xV };
    for (int m = 0; m < 3; ++m) {
        Bump tb = bp;
        float* h1 = (float*)tb.take((size_t)MT * 128 * 4);
        gemm(stream, tb, xin[m], inp[m].w1, inp[m].b1, h1, MT, 128, 3, false, 1);
        gelu_kernel<<<gblk(MT * 128, 256), 256, 0, stream>>>(h1, h1, MT * 128);
        gemm(stream, tb, h1, inp[m].w2, inp[m].b2, xs[m], MT, 128, 128, false, 1);
    }

    // ---- encoder stages ----
    float* gPrev = nullptr;
    for (int s = 0; s < 3; ++s) {
        float* fused = gbuf[s & 1];
        for (int m = 0; m < 3; ++m) {
            const Blk& e = enc[m][s];
            Bump tb = bp;
            bcast_lat_kernel<<<gblk(MLT * DLAT, 256), 256, 0, stream>>>(e.latents, lat);

            float* lq  = (float*)tb.take((size_t)MLT * DLAT * 4);
            float* lc  = (float*)tb.take((size_t)MT * INDIM * 4);
            float* att = (float*)tb.take((size_t)MLT * DLAT * 4);
            ln(stream, lat, e.ca_ln, lq, MLT, DLAT);
            ln(stream, xs[m], e.ca_lnc, lc, MT, INDIM);
            run_attn(stream, tb, lq, NLAT, DLAT, lc, NTOK, INDIM, CH_, CDH_, e.ca, att);
            addv(stream, lat, att, lat, MLT * DLAT);

            if (gPrev) {   // ICMR residual from pooled global latent
                float* gm = (float*)tb.take((size_t)BATCH * DLAT * 4);
                mean_rows_kernel<<<gblk(BATCH * DLAT, 256), 256, 0, stream>>>(gPrev, gm, (int)TFU, DLAT);
                float* proj = (float*)tb.take((size_t)BATCH * NLAT * DLAT * 4);
                gemm(stream, tb, gm, g2l[m][s].w, g2l[m][s].b, proj,
                     BATCH, NLAT * DLAT, DLAT, false, 1);
                addv(stream, lat, proj, lat, MLT * DLAT);
            }

            float* xn = (float*)tb.take((size_t)MLT * DLAT * 4);
            float* a2 = (float*)tb.take((size_t)MLT * DLAT * 4);
            ln(stream, lat, e.sa_ln, xn, MLT, DLAT);
            run_attn(stream, tb, xn, NLAT, DLAT, xn, NLAT, DLAT, SH_, SDH_, e.sa, a2);
            addv(stream, lat, a2, lat, MLT * DLAT);

            float* xf = (float*)tb.take((size_t)MLT * DLAT * 4);
            float* fo = (float*)tb.take((size_t)MLT * DLAT * 4);
            ln(stream, lat, e.ff_ln, xf, MLT, DLAT);
            run_ff(stream, tb, xf, e.ff, MLT, DLAT, fo);
            addv(stream, lat, fo, lat, MLT * DLAT);

            concat_lat_kernel<<<gblk(MLT * DLAT, 256), 256, 0, stream>>>(lat, fused, m);
        }
        // trunk over fused [B, 384, DLAT]
        const long MF = (long)BATCH * TFU;
        for (int t = 0; t < 3; ++t) {
            Bump tb = bp;
            float* xn  = (float*)tb.take((size_t)MF * DLAT * 4);
            float* att = (float*)tb.take((size_t)MF * DLAT * 4);
            ln(stream, fused, trunk[t].sa_ln, xn, MF, DLAT);
            run_attn(stream, tb, xn, TFU, DLAT, xn, TFU, DLAT, SH_, SDH_, trunk[t].sa, att);
            addv(stream, fused, att, fused, MF * DLAT);
            float* xf = (float*)tb.take((size_t)MF * DLAT * 4);
            float* fo = (float*)tb.take((size_t)MF * DLAT * 4);
            ln(stream, fused, trunk[t].ff_ln, xf, MF, DLAT);
            run_ff(stream, tb, xf, trunk[t].ff, MF, DLAT, fo);
            addv(stream, fused, fo, fused, MF * DLAT);
        }
        gPrev = fused;
    }

    // ---- decoders ----
    for (int m = 0; m < 3; ++m) {
        const Dec& d = dec[m];
        Bump tb = bp;
        float* qln = (float*)tb.take((size_t)MQ * QDIM * 4);
        float* cln = (float*)tb.take((size_t)BATCH * TFU * DLAT * 4);
        float* att = (float*)tb.take((size_t)MQ * QDIM * 4);
        ln(stream, queries, d.ca_ln, qln, MQ, QDIM);
        ln(stream, gPrev, d.ca_lnc, cln, (long)BATCH * TFU, DLAT);
        run_attn(stream, tb, qln, NQTOK, QDIM, cln, TFU, DLAT, CH_, CDH_, d.ca, att);

        float* x = (float*)tb.take((size_t)MQ * QDIM * 4);
        addv(stream, x, att, queries, MQ * QDIM);
        float* xn = (float*)tb.take((size_t)MQ * QDIM * 4);
        float* fo = (float*)tb.take((size_t)MQ * QDIM * 4);
        ln(stream, x, d.ff_ln, xn, MQ, QDIM);
        run_ff(stream, tb, xn, d.ff, MQ, QDIM, fo);
        addv(stream, x, x, fo, MQ * QDIM);

        head_kernel<<<gblk(MQ, 8), 256, 0, stream>>>(x, d.hw, d.hb,
            (float*)d_out + (long)m * MQ, (int)MQ, QDIM);
    }
}
